// MultiHeadAttention_43568148251309
// MI455X (gfx1250) — matrix operand order repaired and checked by reading, not compiled, not run on hardware
//
#include <hip/hip_runtime.h>
#include <hip/hip_bf16.h>

// ---------------------------------------------------------------------------
// MI455X (gfx1250) multi-head attention forward, bf16 WMMA path with
// fragment-ready packed operand layouts (per-lane-contiguous 32B loads).
// B=2, T=2048, C=1024, H=16, D=64.
// d_out = [out (B*T*C f32)] ++ [avg_probs (B*T*T f32)]
// ---------------------------------------------------------------------------

typedef __attribute__((ext_vector_type(16))) __bf16 v16bf;
typedef __attribute__((ext_vector_type(8)))  float  v8f;

#define BATCH 2
#define SEQ   2048
#define CDIM  1024
#define NH    16
#define HD    64
#define MROWS (BATCH * SEQ)   // 4096
#define TILE_ELEMS 512        // one 16x32 (A) or 32x16 (B) bf16 tile

// --------------------------- WMMA + fragment helpers ------------------------

__device__ __forceinline__ v8f wmma_bf16(v16bf a, v16bf b, v8f c) {
  return __builtin_amdgcn_wmma_f32_16x16x32_bf16(
      /*neg_a=*/false, a, /*neg_b=*/false, b,
      /*c_mod=*/(short)0, c, /*reuse_a=*/false, /*reuse_b=*/false);
}

// Packed tiles: lane L's 16 bf16 fragment elements are contiguous at
// tile + L*16 elements.  One aligned 32B load per lane.
__device__ __forceinline__ v16bf load_frag(const __bf16* tile) {
  return *(const v16bf*)(tile + (threadIdx.x & 31) * 16);
}

// Packed-A index of element (m=lm 0..15, k=lk 0..31) within a 16x32 tile.
// ISA 7.12.2 A layout: lane = ((lk>>3)&1)*16 + lm ; in-lane order is
// row[kb..kb+7] then row[kb+16..kb+23].
__device__ __forceinline__ int a_pack_idx(int lm, int lk) {
  int lane = ((lk >> 3) & 1) * 16 + lm;
  int j    = (lk & 7) + ((lk >> 4) & 1) * 8;
  return lane * 16 + j;
}

// Packed-B index of element (k=lk 0..31, n=ln 0..15) within a 32x16 tile.
// ISA 7.12.2 B layout: lane = ((lk>>3)&1)*16 + ln ; in-lane order is
// k[kb..kb+7] then k[kb+16..kb+23].
__device__ __forceinline__ int b_pack_idx(int lk, int ln) {
  int lane = ((lk >> 3) & 1) * 16 + ln;
  int j    = (lk & 7) + ((lk >> 4) & 1) * 8;
  return lane * 16 + j;
}

// A fragment from f32 LDS row-major tile (probs), converting to bf16.
// Per lane: two aligned 32B v8f LDS loads.
__device__ __forceinline__ v16bf load_a_f32(const float* p, int ld) {
  int lane = threadIdx.x & 31;
  int m  = lane & 15;
  int kb = (lane >> 4) << 3;
  const float* row = p + m * ld;
  v8f lo = *(const v8f*)(row + kb);
  v8f hi = *(const v8f*)(row + kb + 16);
  v16bf a;
#pragma unroll
  for (int i = 0; i < 8; ++i) {
    a[i]     = (__bf16)lo[i];
    a[8 + i] = (__bf16)hi[i];
  }
  return a;
}

// --------------------------- pack/convert kernels ---------------------------

// x [M][C] f32 -> packed-A bf16 tiles, tile order [mtile][kt] (k-major).
// One thread per packed element: coalesced writes, chunked reads.
__global__ void pack_a_kernel(const float* __restrict__ src,
                              __bf16* __restrict__ dst) {
  int p = blockIdx.x * blockDim.x + threadIdx.x;     // over M*C
  int tile = p >> 9, w = p & 511;
  int lane = w >> 4, j = w & 15;
  int mtile = tile >> 5, kt = tile & 31;             // C/32 = 32 k-tiles
  int lm = lane & 15, hl = lane >> 4;
  int k = kt * 32 + (j >> 3) * 16 + hl * 8 + (j & 7);
  int m = mtile * 16 + lm;
  dst[p] = (__bf16)src[(size_t)m * CDIM + k];
}

// W [N][K] f32 (torch Linear, y = x @ W.T) -> packed-B bf16 tiles of
// B[k][n] = W[n][k], tile order [ntile][kt].
__global__ void pack_w_kernel(const float* __restrict__ w,
                              __bf16* __restrict__ dst) {
  int p = blockIdx.x * blockDim.x + threadIdx.x;     // over C*C
  int tile = p >> 9, ww = p & 511;
  int lane = ww >> 4, j = ww & 15;
  int nt = tile >> 5, kt = tile & 31;
  int ln = lane & 15, khalf = lane >> 4;
  int k = kt * 32 + (j >> 3) * 16 + khalf * 8 + (j & 7);
  int n = nt * 16 + ln;
  dst[p] = (__bf16)w[(size_t)n * CDIM + k];
}

// --------------------------- projection GEMM --------------------------------
// C[m,n] = sum_k A[m,k]*B[k,n] + bias[n], operands fragment-ready in global.
// Block = 4 waves; wave wv owns m-tile (blockIdx.y*4+wv), 4 n-subtiles.
// mode 0: Q  -> packed-A tiles per (b,h): [trow][kt], scaled by 1/8
// mode 1: K  -> packed-B tiles per (b,h): [stile][kt]   (B of Q@K^T)
// mode 2: V  -> packed-B tiles per (b,h): [kt_s][ntile_d] (B of P@V)
// mode 3: f32 out [M][C]
__global__ void gemm_packed_kernel(const __bf16* __restrict__ Apk,
                                   const __bf16* __restrict__ Bpk,
                                   const float* __restrict__ bias, int mode,
                                   __bf16* __restrict__ out_bf,
                                   float* __restrict__ out_f) {
  const int lane  = threadIdx.x & 31;
  const int wv    = threadIdx.x >> 5;                // 0..3
  const int mtile = blockIdx.y * 4 + wv;             // 0..255
  const int nt0   = blockIdx.x * 4;                  // n-tile base (0..63)

  const __bf16* arow = Apk + (size_t)mtile * 32 * TILE_ELEMS;

  v8f acc[4] = {v8f{}, v8f{}, v8f{}, v8f{}};
#pragma unroll 4
  for (int kt = 0; kt < 32; ++kt) {
    __builtin_prefetch(arow + (kt + 1) * TILE_ELEMS + lane * 16, 0, 3);
    v16bf a = load_frag(arow + kt * TILE_ELEMS);
#pragma unroll
    for (int nt = 0; nt < 4; ++nt) {
      v16bf b = load_frag(Bpk + ((size_t)(nt0 + nt) * 32 + kt) * TILE_ELEMS);
      acc[nt] = wmma_bf16(a, b, acc[nt]);
    }
  }

  const int sm = lane & 15;
  const int hl = lane >> 4;
#pragma unroll
  for (int nt = 0; nt < 4; ++nt) {
#pragma unroll
    for (int v = 0; v < 8; ++v) {
      int mg = mtile * 16 + 8 * hl + v;
      int ng = (nt0 + nt) * 16 + sm;
      float val = acc[nt][v] + bias[ng];
      int bI = mg >> 11;                // / SEQ
      int t  = mg & (SEQ - 1);
      int h  = ng >> 6;                 // / HD
      int d  = ng & (HD - 1);
      size_t bh = (size_t)(bI * NH + h);
      if (mode == 0) {                  // Q: packed-A [trow][kt], scale 1/8
        size_t idx = ((bh * (SEQ / 16) + (t >> 4)) * 2 + (d >> 5)) * TILE_ELEMS
                   + a_pack_idx(t & 15, d & 31);
        out_bf[idx] = (__bf16)(val * 0.125f);
      } else if (mode == 1) {           // K: packed-B [stile][kt]
        size_t idx = ((bh * (SEQ / 16) + (t >> 4)) * 2 + (d >> 5)) * TILE_ELEMS
                   + b_pack_idx(d & 31, t & 15);
        out_bf[idx] = (__bf16)val;
      } else if (mode == 2) {           // V: packed-B [kt_s][ntile_d]
        size_t idx = ((bh * (SEQ / 32) + (t >> 5)) * 4 + (d >> 4)) * TILE_ELEMS
                   + b_pack_idx(t & 31, d & 15);
        out_bf[idx] = (__bf16)val;
      } else {                          // final projection, f32
        out_f[(size_t)mg * CDIM + ng] = val;
      }
    }
  }
}

// --------------------------- attention kernel -------------------------------
// One workgroup (16 waves / 512 threads) per (batch, 16-query-row block).
// LDS: S/probs 16x2048 f32 (128KB) + avgP 16x2048 f32 (128KB) + O 16x64 f32
//      = 266,240 B (< 320KB/WGP).  Q/K/V fragments stream from packed global.
#define ATTN_SMEM_BYTES (16 * SEQ * 4 * 2 + 16 * HD * 4)

__global__ void attention_kernel(const __bf16* __restrict__ qpk,  // packed-A
                                 const __bf16* __restrict__ kpk,  // packed-B
                                 const __bf16* __restrict__ vpk,  // packed-B
                                 __bf16* __restrict__ apk,        // packed-A out
                                 float* __restrict__ avg_out) {   // [B,T,T]
  extern __shared__ char smem[];
  float* S  = (float*)smem;             // [16][SEQ]  scores -> probs
  float* AP = S + 16 * SEQ;             // [16][SEQ]  avg-prob accumulator
  float* OS = AP + 16 * SEQ;            // [16][HD]   O reduction

  const int tid  = threadIdx.x;
  const int lane = tid & 31;
  const int wv   = tid >> 5;            // 0..15
  const int b    = blockIdx.x >> 7;     // / (SEQ/16)
  const int tb   = blockIdx.x & 127;    // query row-tile within batch
  const int t0   = tb * 16;
  const int tmax = t0 + 15;
  const int sm   = lane & 15;
  const int hl   = lane >> 4;

  for (int i = tid; i < 16 * SEQ; i += 512) AP[i] = 0.0f;
  __syncthreads();

  for (int h = 0; h < NH; ++h) {
    const size_t bh = (size_t)(b * NH + h);
    const __bf16* qh = qpk + (bh * (SEQ / 16) + tb) * 2 * TILE_ELEMS;
    const __bf16* kh = kpk + bh * (SEQ / 16) * 2 * TILE_ELEMS;
    const __bf16* vh = vpk + bh * (SEQ / 32) * 4 * TILE_ELEMS;

    for (int i = tid; i < 16 * HD; i += 512) OS[i] = 0.0f;

    // Q fragments for this head (K-dim = 64 -> two k-tiles), from global.
    v16bf aq0 = load_frag(qh);
    v16bf aq1 = load_frag(qh + TILE_ELEMS);
    __syncthreads();

    // ---- S = Qs @ K^T : wave wv owns columns [wv*128, wv*128+128)
    for (int st = 0; st < 8; ++st) {
      int s0 = wv * 128 + st * 16;
      if (s0 > tmax) {                  // fully causal-masked tile
#pragma unroll
        for (int v = 0; v < 8; ++v)
          S[(8 * hl + v) * SEQ + s0 + sm] = -__builtin_inff();
        continue;
      }
      const __bf16* kt_base = kh + (size_t)(s0 >> 4) * 2 * TILE_ELEMS;
      v8f acc = {};
      acc = wmma_bf16(aq0, load_frag(kt_base), acc);
      acc = wmma_bf16(aq1, load_frag(kt_base + TILE_ELEMS), acc);
#pragma unroll
      for (int v = 0; v < 8; ++v) {
        int mr = 8 * hl + v;
        int sg = s0 + sm;
        S[mr * SEQ + sg] = (sg <= t0 + mr) ? acc[v] : -__builtin_inff();
      }
    }
    __syncthreads();

    // ---- softmax: wave wv handles row wv; accumulate avg probs
    {
      const int r = wv;
      float* Sr = S + r * SEQ;
      float mx = -__builtin_inff();
      for (int s = lane; s < SEQ; s += 32) mx = fmaxf(mx, Sr[s]);
#pragma unroll
      for (int off = 16; off > 0; off >>= 1)
        mx = fmaxf(mx, __shfl_xor(mx, off, 32));
      float sum = 0.0f;
      for (int s = lane; s < SEQ; s += 32) {
        float e = __expf(Sr[s] - mx);   // exp(-inf - mx) == 0
        Sr[s] = e;
        sum += e;
      }
#pragma unroll
      for (int off = 16; off > 0; off >>= 1)
        sum += __shfl_xor(sum, off, 32);
      float inv = 1.0f / sum;
      float* APr = AP + r * SEQ;
      for (int s = lane; s < SEQ; s += 32) {
        float p = Sr[s] * inv;
        Sr[s] = p;
        APr[s] += p * (1.0f / NH);
      }
    }
    __syncthreads();

    // ---- O_h = P @ V : wave wv reduces its 128 columns, LDS f32 atomics
    if (wv * 128 <= tmax) {
      v8f oacc[4] = {v8f{}, v8f{}, v8f{}, v8f{}};
      for (int ks = 0; ks < 4; ++ks) {
        int s0 = wv * 128 + ks * 32;
        if (s0 > tmax) break;           // probs are zero past here
        v16bf afrag = load_a_f32(S + s0, SEQ);
        const __bf16* vt = vh + (size_t)(s0 >> 5) * 4 * TILE_ELEMS;
#pragma unroll
        for (int nt = 0; nt < 4; ++nt)
          oacc[nt] = wmma_bf16(afrag, load_frag(vt + nt * TILE_ELEMS),
                               oacc[nt]);
      }
#pragma unroll
      for (int nt = 0; nt < 4; ++nt)
#pragma unroll
        for (int v = 0; v < 8; ++v)
          atomicAdd(&OS[(8 * hl + v) * HD + nt * 16 + sm], oacc[nt][v]);
    }
    __syncthreads();

    // ---- store O tile (bf16) as packed-A for the output projection
    {
      const int mtile0 = b * (SEQ / 16) + tb;       // global m-tile
      for (int i = tid; i < 16 * HD; i += 512) {
        int mr = i >> 6, c = i & (HD - 1);
        int kt = h * 2 + (c >> 5);
        apk[((size_t)mtile0 * 32 + kt) * TILE_ELEMS + a_pack_idx(mr, c & 31)] =
            (__bf16)OS[i];
      }
    }
    __syncthreads();
  }

  // ---- write avg_probs block [16 rows x SEQ]
  for (int i = tid; i < 16 * SEQ; i += 512) {
    int m = i >> 11, s = i & (SEQ - 1);
    avg_out[((size_t)(b * SEQ) + t0 + m) * SEQ + s] = AP[i];
  }
}

// --------------------------- launch -----------------------------------------

extern "C" void kernel_launch(void* const* d_in, const int* in_sizes, int n_in,
                              void* d_out, int out_size, void* d_ws, size_t ws_size,
                              hipStream_t stream) {
  (void)in_sizes; (void)n_in; (void)out_size; (void)ws_size;

  const float* x  = (const float*)d_in[0];
  const float* Wq = (const float*)d_in[1];
  const float* bq = (const float*)d_in[2];
  const float* Wk = (const float*)d_in[3];
  const float* bk = (const float*)d_in[4];
  const float* Wv = (const float*)d_in[5];
  const float* bv = (const float*)d_in[6];
  const float* Wo = (const float*)d_in[7];
  const float* bo = (const float*)d_in[8];

  float* out = (float*)d_out;                       // [B,T,C]
  float* avg = out + (size_t)MROWS * CDIM;          // [B,T,T]

  char* ws = (char*)d_ws;
  __bf16* xpk  = (__bf16*)(ws);                     //  8 MB packed-A x
  __bf16* wtq  = (__bf16*)(ws + (8ull  << 20));     //  2 MB packed-B Wq
  __bf16* wtk  = (__bf16*)(ws + (10ull << 20));     //  2 MB packed-B Wk
  __bf16* wtv  = (__bf16*)(ws + (12ull << 20));     //  2 MB packed-B Wv
  __bf16* wto  = (__bf16*)(ws + (14ull << 20));     //  2 MB packed-B Wo
  __bf16* qpk  = (__bf16*)(ws + (16ull << 20));     //  8 MB packed-A Q (scaled)
  __bf16* kpk  = (__bf16*)(ws + (24ull << 20));     //  8 MB packed-B K
  __bf16* vpk  = (__bf16*)(ws + (32ull << 20));     //  8 MB packed-B V
  __bf16* apk  = (__bf16*)(ws + (40ull << 20));     //  8 MB packed-A attn-out

  // 1) pack fp32 inputs into fragment-ready bf16 tile layouts
  pack_a_kernel<<<(MROWS * CDIM) / 256, 256, 0, stream>>>(x, xpk);
  pack_w_kernel<<<(CDIM * CDIM) / 256, 256, 0, stream>>>(Wq, wtq);
  pack_w_kernel<<<(CDIM * CDIM) / 256, 256, 0, stream>>>(Wk, wtk);
  pack_w_kernel<<<(CDIM * CDIM) / 256, 256, 0, stream>>>(Wv, wtv);
  pack_w_kernel<<<(CDIM * CDIM) / 256, 256, 0, stream>>>(Wo, wto);

  // 2) Q/K/V projections (bf16 WMMA, outputs re-packed for attention)
  dim3 ggrid(CDIM / 64, MROWS / 64);   // (16, 64)
  gemm_packed_kernel<<<ggrid, 128, 0, stream>>>(xpk, wtq, bq, 0, qpk, nullptr);
  gemm_packed_kernel<<<ggrid, 128, 0, stream>>>(xpk, wtk, bk, 1, kpk, nullptr);
  gemm_packed_kernel<<<ggrid, 128, 0, stream>>>(xpk, wtv, bv, 2, vpk, nullptr);

  // 3) attention (scores + softmax + avg_probs + P@V) — 260KB LDS/WG
  hipFuncSetAttribute((const void*)attention_kernel,
                      hipFuncAttributeMaxDynamicSharedMemorySize,
                      ATTN_SMEM_BYTES);
  attention_kernel<<<BATCH * (SEQ / 16), 512, ATTN_SMEM_BYTES, stream>>>(
      qpk, kpk, vpk, apk, avg);

  // 4) output projection -> f32 d_out
  gemm_packed_kernel<<<ggrid, 128, 0, stream>>>(apk, wto, bo, 3, nullptr, out);
}
